// Llama3MHAttention_56212531970357
// MI455X (gfx1250) — matrix operand order repaired and checked by reading, not compiled, not run on hardware
//
#include <hip/hip_runtime.h>
#include <hip/hip_bf16.h>
#include <math.h>

// ---------------------------------------------------------------------------
// Types for CDNA5 WMMA (wave32)
// ---------------------------------------------------------------------------
typedef _Float16 v8h  __attribute__((ext_vector_type(8)));
typedef _Float16 v16h __attribute__((ext_vector_type(16)));
typedef float    v8f  __attribute__((ext_vector_type(8)));
typedef int      v4i  __attribute__((ext_vector_type(4)));

#define D_MODEL 4096
#define N_HEADS 32
#define N_KV    8
#define D_HEAD  128
#define SEQ_T   2048
#define KV_DIM  (N_KV * D_HEAD)   // 1024

// Async global->LDS support (gfx1250). Guarded so the build cannot break.
#ifndef HAVE_ASYNC_LDS
#  if defined(__has_builtin)
#    if __has_builtin(__builtin_amdgcn_global_load_async_to_lds_b128)
#      define HAVE_ASYNC_LDS 1
#    else
#      define HAVE_ASYNC_LDS 0
#    endif
#  else
#    define HAVE_ASYNC_LDS 0
#  endif
#endif

#if HAVE_ASYNC_LDS
typedef __attribute__((address_space(1))) v4i g_v4i;   // global int4
typedef __attribute__((address_space(3))) v4i l_v4i;   // LDS int4
#endif

__device__ __forceinline__ v8f wmma_f16(v16h a, v16h b, v8f c) {
    // D = A(16x32 f16) * B(32x16 f16) + C(16x16 f32)
    return __builtin_amdgcn_wmma_f32_16x16x32_f16(
        /*neg_a=*/false, a, /*neg_b=*/false, b,
        /*c_mod=*/(short)0, c, /*reuse_a=*/false, /*reuse_b=*/false);
}

__device__ __forceinline__ float red_max16(float v) {
    v = fmaxf(v, __shfl_xor(v, 1, 32));
    v = fmaxf(v, __shfl_xor(v, 2, 32));
    v = fmaxf(v, __shfl_xor(v, 4, 32));
    v = fmaxf(v, __shfl_xor(v, 8, 32));
    return v;
}
__device__ __forceinline__ float red_sum16(float v) {
    v += __shfl_xor(v, 1, 32);
    v += __shfl_xor(v, 2, 32);
    v += __shfl_xor(v, 4, 32);
    v += __shfl_xor(v, 8, 32);
    return v;
}

// ---------------------------------------------------------------------------
// 1) fp32 -> f16 convert
// ---------------------------------------------------------------------------
__global__ void cvt_f32_to_f16(const float* __restrict__ in,
                               _Float16* __restrict__ out, int n) {
    int i = blockIdx.x * blockDim.x + threadIdx.x;
    if (i < n) out[i] = (_Float16)in[i];
}

// ---------------------------------------------------------------------------
// 2) W[K x N] fp32 row-major -> WT[N x K] f16 (K-contiguous rows for B operand)
// ---------------------------------------------------------------------------
__global__ void transpose_w_f16(const float* __restrict__ W,
                                _Float16* __restrict__ WT, int K, int N) {
    int idx = blockIdx.x * blockDim.x + threadIdx.x;
    if (idx >= K * N) return;
    int r = idx / N;     // input-dim index
    int c = idx % N;     // output-col index
    WT[(size_t)c * K + r] = (_Float16)W[idx];
}

// half R x C -> half C x R (for V -> V^T)
__global__ void transpose_h(const _Float16* __restrict__ in,
                            _Float16* __restrict__ out, int R, int C) {
    int idx = blockIdx.x * blockDim.x + threadIdx.x;
    if (idx >= R * C) return;
    int r = idx / C;
    int c = idx % C;
    out[(size_t)c * R + r] = in[idx];
}

// ---------------------------------------------------------------------------
// 3) Tiled GEMM: C[M x N] = A[M x Kd] f16 * BT[N x Kd]^T, fp32 accumulate.
//    256 threads (8 waves) per block; block tile 128(M) x 128(N).
//    Wave w owns rows [m0 + 16w, +16); all 8 waves share the B tile, which is
//    double-buffered in LDS and filled with async global->LDS loads.
// ---------------------------------------------------------------------------
#define BROW_H 40   // LDS row pitch in halfs (32 data + 8 pad, 16B-aligned rows)

// stage 128 rows x 32 halfs of BT (cols k..k+31) into LDS buffer
__device__ __forceinline__ void stage_b(const _Float16* __restrict__ BT,
                                        _Float16* dst, int n0, int k, int Kd,
                                        int tid) {
    int row = tid >> 1;          // 0..127
    int seg = tid & 1;           // which 32B segment of the 64B row-chunk
    const _Float16* g = BT + (size_t)(n0 + row) * Kd + k + seg * 16;
    _Float16* l = dst + row * BROW_H + seg * 16;
#if HAVE_ASYNC_LDS
    __builtin_amdgcn_global_load_async_to_lds_b128((g_v4i*)g, (l_v4i*)l, 0, 0);
    __builtin_amdgcn_global_load_async_to_lds_b128((g_v4i*)(g + 8), (l_v4i*)(l + 8), 0, 0);
#else
    *(v8h*)l = *(const v8h*)g;
    *(v8h*)(l + 8) = *(const v8h*)(g + 8);
#endif
}

template <typename OutT>
__global__ __launch_bounds__(256)
void gemm_tiled(const _Float16* __restrict__ A, const _Float16* __restrict__ BT,
                OutT* __restrict__ C, int M, int N, int Kd) {
    __shared__ __align__(32) _Float16 bsh[2][128 * BROW_H];   // 2 x 10KB

    const int tid  = threadIdx.x;
    const int lane = tid & 31;
    const int wid  = tid >> 5;           // 0..7
    const int l16  = lane & 15;
    const int lh   = lane >> 4;          // 0 or 1 (lane half)
    const int m0   = blockIdx.x * 128 + wid * 16;
    const int n0   = blockIdx.y * 128;

    v8f z = {0,0,0,0,0,0,0,0};
    v8f acc[8] = {z, z, z, z, z, z, z, z};

    const _Float16* arow = A + (size_t)(m0 + l16) * Kd;
    const int nk = Kd / 32;

    stage_b(BT, &bsh[0][0], n0, 0, Kd, tid);
    int cur = 0;
    for (int i = 0; i < nk; ++i) {
        const int k = i * 32;
        const int knext = (i + 1 < nk) ? k + 32 : k;   // clamp (harmless dup)
        stage_b(BT, &bsh[cur ^ 1][0], n0, knext, Kd, tid);
#if HAVE_ASYNC_LDS
        asm volatile("s_wait_asynccnt 0x2" ::: "memory");
#endif
        __syncthreads();

        v16h a;
        ((v8h*)&a)[0] = *(const v8h*)(arow + k + lh * 16);
        ((v8h*)&a)[1] = *(const v8h*)(arow + k + lh * 16 + 8);

        const _Float16* bbase = &bsh[cur][0];
        #pragma unroll
        for (int j = 0; j < 8; ++j) {
            const _Float16* bp = bbase + (j * 16 + l16) * BROW_H + lh * 16;
            v16h b;
            ((v8h*)&b)[0] = *(const v8h*)(bp);
            ((v8h*)&b)[1] = *(const v8h*)(bp + 8);
            acc[j] = wmma_f16(a, b, acc[j]);
        }
        __syncthreads();   // everyone done reading bsh[cur] before re-staging
        cur ^= 1;
    }

    #pragma unroll
    for (int j = 0; j < 8; ++j)
        #pragma unroll
        for (int r = 0; r < 8; ++r)
            C[(size_t)(m0 + r + 8 * lh) * N + n0 + j * 16 + l16] =
                (OutT)acc[j][r];
}

// ---------------------------------------------------------------------------
// 4) Llama3-scaled RoPE, in place on f16 [T, heads*128]
// ---------------------------------------------------------------------------
__global__ void rope_f16(_Float16* __restrict__ buf, int heads, int stride) {
    int idx = blockIdx.x * blockDim.x + threadIdx.x;
    int total = SEQ_T * heads * 64;
    if (idx >= total) return;
    int i = idx % 64;
    int h = (idx / 64) % heads;
    int t = idx / (64 * heads);

    const float TWO_PI = 6.283185307179586f;
    float freq = (1.0f / TWO_PI) * __expf(-((float)i / 64.0f) * logf(500000.0f));
    const float L0 = 8192.0f, factor = 32.0f, lo = 1.0f, hi = 4.0f;
    float freq_low  = lo / L0;
    float freq_high = hi / L0;
    float freq_scaled = (freq < freq_low) ? freq / factor : freq;
    float smooth = fminf(fmaxf((L0 * freq - lo) / (hi - lo), 0.0f), 1.0f);
    float freq_smooth = (1.0f - smooth) * (freq / factor) + smooth * freq;
    bool  is_mid = (freq >= freq_low) && (freq <= freq_high);
    float f = is_mid ? freq_smooth : freq_scaled;

    float phase = TWO_PI * (float)t * f;
    float c, s;
    __sincosf(phase, &s, &c);

    size_t base = (size_t)t * stride + h * D_HEAD;
    float x0 = (float)buf[base + i];
    float x1 = (float)buf[base + i + 64];
    buf[base + i]      = (_Float16)(x0 * c - x1 * s);
    buf[base + i + 64] = (_Float16)(x1 * c + x0 * s);
}

// ---------------------------------------------------------------------------
// 5) Flash attention: one wave per (head, 16-row q tile), online softmax.
//    Q:  [T, 4096] f16 (post-RoPE), head h cols h*128..h*128+127
//    Kb: [T, 1024] f16 (post-RoPE), kv head cols kvh*128..
//    VT: [1024, T] f16 (d-major, keys contiguous)
//    ctx:[T, 4096] f16
// ---------------------------------------------------------------------------
__global__ __launch_bounds__(32)
void flash_attn(const _Float16* __restrict__ Q, const _Float16* __restrict__ Kb,
                const _Float16* __restrict__ VT, _Float16* __restrict__ ctx) {
    const int lane = threadIdx.x;
    const int l16  = lane & 15;
    const int lh   = lane >> 4;
    const int q0   = blockIdx.x * 16;
    const int h    = blockIdx.y;
    const int kvh  = h >> 2;
    const float sc = 0.08838834764831845f;   // 1/sqrt(128)

    __shared__ __align__(32) _Float16 plds[16 * 32];

    // Preload Q tile in A layout: 4 chunks of K=32 covering d=0..127
    v16h qa[4];
    const _Float16* qbase = Q + (size_t)(q0 + l16) * D_MODEL + h * D_HEAD;
    #pragma unroll
    for (int c = 0; c < 4; ++c) {
        v16h t;
        ((v8h*)&t)[0] = *(const v8h*)(qbase + c * 32 + lh * 16);
        ((v8h*)&t)[1] = *(const v8h*)(qbase + c * 32 + lh * 16 + 8);
        qa[c] = t;
    }

    v8f z = {0,0,0,0,0,0,0,0};
    v8f acc[8] = {z, z, z, z, z, z, z, z};
    float mrun[8], lrun[8];
    #pragma unroll
    for (int r = 0; r < 8; ++r) { mrun[r] = -3.0e38f; lrun[r] = 0.0f; }

    const int kend = q0 + 16;   // causal: keys 0..q0+15
    for (int kb = 0; kb < kend; kb += 32) {
        // ---- S = Q * K^T for 32 keys (two 16-key tiles) ----
        v8f s0 = z, s1 = z;
        const _Float16* kr0 = Kb + (size_t)(kb + l16) * KV_DIM
                                 + kvh * D_HEAD + lh * 16;
        const _Float16* kr1 = kr0 + (size_t)16 * KV_DIM;
        #pragma unroll
        for (int c = 0; c < 4; ++c) {
            v16h b0 = *(const v16h*)(kr0 + c * 32);
            s0 = wmma_f16(qa[c], b0, s0);
        }
        #pragma unroll
        for (int c = 0; c < 4; ++c) {
            v16h b1 = *(const v16h*)(kr1 + c * 32);
            s1 = wmma_f16(qa[c], b1, s1);
        }

        // ---- masked online softmax (fp32) ----
        float scale[8], p0[8], p1[8];
        #pragma unroll
        for (int r = 0; r < 8; ++r) {
            int rowg = q0 + r + 8 * lh;
            float a0 = s0[r] * sc;
            float a1 = s1[r] * sc;
            if (kb + l16 > rowg)      a0 = -3.0e38f;
            if (kb + 16 + l16 > rowg) a1 = -3.0e38f;
            float mx = red_max16(fmaxf(a0, a1));
            float mnew = fmaxf(mrun[r], mx);
            p0[r] = __expf(a0 - mnew);
            p1[r] = __expf(a1 - mnew);
            float su = red_sum16(p0[r] + p1[r]);
            scale[r] = __expf(mrun[r] - mnew);
            lrun[r]  = lrun[r] * scale[r] + su;
            mrun[r]  = mnew;
        }
        #pragma unroll
        for (int nd = 0; nd < 8; ++nd)
            #pragma unroll
            for (int r = 0; r < 8; ++r)
                acc[nd][r] *= scale[r];

        // ---- restage P (16x32) through LDS into A layout ----
        #pragma unroll
        for (int r = 0; r < 8; ++r) {
            plds[(r + 8 * lh) * 32 + l16]      = (_Float16)p0[r];
            plds[(r + 8 * lh) * 32 + 16 + l16] = (_Float16)p1[r];
        }
        asm volatile("s_wait_dscnt 0x0" ::: "memory");
        v16h pa;
        const _Float16* pr = plds + l16 * 32 + lh * 16;
        ((v8h*)&pa)[0] = *(const v8h*)(pr);
        ((v8h*)&pa)[1] = *(const v8h*)(pr + 8);

        // ---- ctx += P * V (B operand from V^T, keys contiguous) ----
        #pragma unroll
        for (int nd = 0; nd < 8; ++nd) {
            v16h b = *(const v16h*)(VT + (size_t)(kvh * D_HEAD + nd * 16 + l16)
                                             * SEQ_T + kb + lh * 16);
            acc[nd] = wmma_f16(pa, b, acc[nd]);
        }
    }

    // ---- normalize + store ctx ----
    #pragma unroll
    for (int nd = 0; nd < 8; ++nd)
        #pragma unroll
        for (int r = 0; r < 8; ++r) {
            float val = acc[nd][r] / lrun[r];
            ctx[(size_t)(q0 + r + 8 * lh) * D_MODEL + h * D_HEAD + nd * 16 + l16]
                = (_Float16)val;
        }
}

// ---------------------------------------------------------------------------
// Host launcher
// ---------------------------------------------------------------------------
extern "C" void kernel_launch(void* const* d_in, const int* in_sizes, int n_in,
                              void* d_out, int out_size, void* d_ws, size_t ws_size,
                              hipStream_t stream) {
    const float* resid = (const float*)d_in[0];
    const float* Wq    = (const float*)d_in[1];
    const float* Wk    = (const float*)d_in[2];
    const float* Wv    = (const float*)d_in[3];
    const float* Wo    = (const float*)d_in[4];
    float* out = (float*)d_out;

    char* ws = (char*)d_ws;
    size_t off = 0;
    auto alloc = [&](size_t bytes) { char* p = ws + off; off += bytes; return p; };

    _Float16* Xh   = (_Float16*)alloc((size_t)SEQ_T * D_MODEL * 2);   // 16 MB
    _Float16* WqT  = (_Float16*)alloc((size_t)D_MODEL * D_MODEL * 2); // 32 MB
    _Float16* WkT  = (_Float16*)alloc((size_t)KV_DIM * D_MODEL * 2);  // 8 MB
    _Float16* WvT  = (_Float16*)alloc((size_t)KV_DIM * D_MODEL * 2);  // 8 MB
    _Float16* WoT  = (_Float16*)alloc((size_t)D_MODEL * D_MODEL * 2); // 32 MB
    _Float16* Qb   = (_Float16*)alloc((size_t)SEQ_T * D_MODEL * 2);   // 16 MB
    _Float16* Kbuf = (_Float16*)alloc((size_t)SEQ_T * KV_DIM * 2);    // 4 MB
    _Float16* Vbuf = (_Float16*)alloc((size_t)SEQ_T * KV_DIM * 2);    // 4 MB
    _Float16* VT   = (_Float16*)alloc((size_t)KV_DIM * SEQ_T * 2);    // 4 MB
    _Float16* Ctx  = (_Float16*)alloc((size_t)SEQ_T * D_MODEL * 2);   // 16 MB
    (void)ws_size; (void)in_sizes; (void)n_in; (void)out_size;

    // 1) resid -> f16
    {
        int n = SEQ_T * D_MODEL;
        cvt_f32_to_f16<<<(n + 255) / 256, 256, 0, stream>>>(resid, Xh, n);
    }
    // 2) weight transposes (fp32 -> f16, K-contiguous rows)
    {
        int n = D_MODEL * D_MODEL;
        transpose_w_f16<<<(n + 255) / 256, 256, 0, stream>>>(Wq, WqT, D_MODEL, D_MODEL);
        transpose_w_f16<<<(n + 255) / 256, 256, 0, stream>>>(Wo, WoT, D_MODEL, D_MODEL);
        int m = D_MODEL * KV_DIM;
        transpose_w_f16<<<(m + 255) / 256, 256, 0, stream>>>(Wk, WkT, D_MODEL, KV_DIM);
        transpose_w_f16<<<(m + 255) / 256, 256, 0, stream>>>(Wv, WvT, D_MODEL, KV_DIM);
    }
    // 3) projections (tiled WMMA, async-LDS double buffered)
    gemm_tiled<_Float16><<<dim3(SEQ_T / 128, D_MODEL / 128), 256, 0, stream>>>(
        Xh, WqT, Qb, SEQ_T, D_MODEL, D_MODEL);
    gemm_tiled<_Float16><<<dim3(SEQ_T / 128, KV_DIM / 128), 256, 0, stream>>>(
        Xh, WkT, Kbuf, SEQ_T, KV_DIM, D_MODEL);
    gemm_tiled<_Float16><<<dim3(SEQ_T / 128, KV_DIM / 128), 256, 0, stream>>>(
        Xh, WvT, Vbuf, SEQ_T, KV_DIM, D_MODEL);
    // 4) RoPE on Q and K
    {
        int nq = SEQ_T * N_HEADS * 64;
        rope_f16<<<(nq + 255) / 256, 256, 0, stream>>>(Qb, N_HEADS, D_MODEL);
        int nk = SEQ_T * N_KV * 64;
        rope_f16<<<(nk + 255) / 256, 256, 0, stream>>>(Kbuf, N_KV, KV_DIM);
    }
    // 5) V -> V^T (keys contiguous per d)
    {
        int n = SEQ_T * KV_DIM;
        transpose_h<<<(n + 255) / 256, 256, 0, stream>>>(Vbuf, VT, SEQ_T, KV_DIM);
    }
    // 6) flash attention (WMMA + online softmax)
    flash_attn<<<dim3(SEQ_T / 16, N_HEADS), 32, 0, stream>>>(Qb, Kbuf, VT, Ctx);
    // 7) output projection -> fp32 d_out
    gemm_tiled<float><<<dim3(SEQ_T / 128, D_MODEL / 128), 256, 0, stream>>>(
        Ctx, WoT, out, SEQ_T, D_MODEL, D_MODEL);
}
